// PoseMSMetaResNet_37933151158850
// MI455X (gfx1250) — matrix operand order repaired and checked by reading, not compiled, not run on hardware
//
#include <hip/hip_runtime.h>

typedef __attribute__((ext_vector_type(16))) _Float16 v16h;
typedef __attribute__((ext_vector_type(8)))  _Float16 v8h;
typedef __attribute__((ext_vector_type(8)))  float    v8f;
typedef __attribute__((ext_vector_type(4)))  int      v4i;

#if __has_builtin(__builtin_amdgcn_global_load_async_to_lds_b128)
#define HAVE_ASYNC_LDS 1
#endif

__device__ __forceinline__ void wait_async0() {
#if __has_builtin(__builtin_amdgcn_s_wait_asynccnt)
    __builtin_amdgcn_s_wait_asynccnt(0);
#else
    asm volatile("s_wait_asynccnt 0x0" ::: "memory");
#endif
}

// ---------------------------------------------------------------- utilities

__global__ void k_tohalf(const float* __restrict__ in, _Float16* __restrict__ out, long n) {
    long i = blockIdx.x * (long)blockDim.x + threadIdx.x;
    if (i < n) out[i] = (_Float16)in[i];
}

// Fold BN (scale/shift) into f16 weights, pad K to Kpad with zeros.
__global__ void k_foldbn(const float* __restrict__ w,
                         const float* __restrict__ g, const float* __restrict__ b,
                         const float* __restrict__ m, const float* __restrict__ v,
                         _Float16* __restrict__ w16, float* __restrict__ bias,
                         int OC, int K, int Kpad) {
    int idx = blockIdx.x * blockDim.x + threadIdx.x;
    if (idx >= OC * Kpad) return;
    int oc = idx / Kpad;
    int k  = idx - oc * Kpad;
    float scale = g[oc] * rsqrtf(v[oc] + 1e-5f);
    if (k == 0) bias[oc] = b[oc] - m[oc] * scale;
    w16[idx] = (_Float16)(k < K ? w[oc * K + k] * scale : 0.f);
}

// ---------------------------------------------------------------- implicit-GEMM conv (WMMA)

struct ConvP {
    const _Float16* in;  const _Float16* w; const float* bias;
    const _Float16* res; _Float16* out;
    int N, IC, IH, IW, OC, OH, OW, KH, KW, st, pad, relu, Kc, Kpad;
};

// block = 128 threads (4 wave32). Wave w computes the 16x16 tile
// (16 output pixels) x (channels blockIdx.y*64 + w*16 .. +15).
// GEMM: M = N*OH*OW, N = OC, K = IC*KH*KW (padded to 32).
__global__ void k_conv_wmma(ConvP p) {
    __shared__ _Float16 lA[16][40];                   // im2col tile, 16-B aligned rows
    const int t = threadIdx.x, lane = t & 31;
    const int mtile  = blockIdx.x;
    const int wave   = t >> 5;
    const int octile = blockIdx.y * 64 + wave * 16;
    const int OHW = p.OH * p.OW;
    const int M   = p.N * OHW;
    const int KHW = p.KH * p.KW;

    // ---- loop-invariant im2col decode for this thread's staging slot
    const int mloc  = t >> 3;            // 0..15
    const int k0    = (t & 7) * 4;       // 0..28
    const int pixel = mtile * 16 + mloc;
    const bool pv   = pixel < M;
    const int pix = pv ? pixel : 0;
    const int img = pix / OHW, hw = pix - img * OHW;
    const int oy = hw / p.OW, ox = hw - oy * p.OW;
    const int iy0 = oy * p.st - p.pad, ix0 = ox * p.st - p.pad;
    const _Float16* __restrict__ inImg = p.in + (long)img * p.IC * p.IH * p.IW;

    const int g = lane >> 4, nn = lane & 15;
    const _Float16* __restrict__ wr0 = p.w + (long)(octile + nn) * p.Kpad + 16 * g;

    v8f acc = {0.f,0.f,0.f,0.f,0.f,0.f,0.f,0.f};

    for (int kbase = 0; kbase < p.Kpad; kbase += 32) {
        // ---- stage 16x32 im2col A-tile: 128 threads x 4 halfs
#pragma unroll
        for (int j = 0; j < 4; j++) {
            int kk = kbase + k0 + j;
            float val = 0.f;
            if (pv && kk < p.Kc) {
                int ic = kk / KHW, r = kk - ic * KHW;
                int kh = r / p.KW, kw = r - kh * p.KW;
                int iy = iy0 + kh, ix = ix0 + kw;
                if (iy >= 0 && iy < p.IH && ix >= 0 && ix < p.IW)
                    val = (float)inImg[((long)ic * p.IH + iy) * p.IW + ix];
            }
            lA[mloc][k0 + j] = (_Float16)val;
        }
        __syncthreads();

        // ---- A fragment: 16 contiguous K halfs per lane, 2x ds_read_b128
        const v8h* rowp = (const v8h*)&lA[nn][0];
        v8h alo = rowp[2 * g], ahi = rowp[2 * g + 1];
        // ---- B fragment: 16 contiguous K halfs per lane, 2x global_load_b128
        const v8h* wp = (const v8h*)(wr0 + kbase);
        __builtin_prefetch(wr0 + kbase + 32, 0, 1);
        v8h blo = wp[0], bhi = wp[1];
        __syncthreads();   // A-tile consumed; safe to restage next chunk

        v16h a, b;
#pragma unroll
        for (int e = 0; e < 8; e++) {
            a[e] = alo[e]; a[8 + e] = ahi[e];
            b[e] = blo[e]; b[8 + e] = bhi[e];
        }
        acc = __builtin_amdgcn_wmma_f32_16x16x32_f16(false, a, false, b,
                                                     (short)0, acc, false, false);
    }

    // ---- epilogue: bias (+residual) (+relu), f16 store
#pragma unroll
    for (int r = 0; r < 8; r++) {
        int ml  = r + 8 * g;
        int px  = mtile * 16 + ml;
        if (px < M) {
            int im = px / OHW, h2 = px - im * OHW;
            int oc = octile + nn;
            long idx = ((long)im * p.OC + oc) * OHW + h2;
            float v = acc[r] + p.bias[oc];
            if (p.res)  v += (float)p.res[idx];
            if (p.relu) v = v > 0.f ? v : 0.f;
            p.out[idx] = (_Float16)v;
        }
    }
}

// ---------------------------------------------------------------- maxpool 3x3 s2 p1

__global__ void k_maxpool(const _Float16* __restrict__ in, _Float16* __restrict__ out,
                          int NC, int IH, int IW, int OH, int OW) {
    int idx = blockIdx.x * blockDim.x + threadIdx.x;
    int total = NC * OH * OW;
    if (idx >= total) return;
    int ow = idx % OW; int tmp = idx / OW;
    int oh = tmp % OH; int nc  = tmp / OH;
    float best = -3.4e38f;
#pragma unroll
    for (int dy = 0; dy < 3; dy++)
#pragma unroll
        for (int dx = 0; dx < 3; dx++) {
            int iy = oh * 2 - 1 + dy, ix = ow * 2 - 1 + dx;
            if (iy >= 0 && iy < IH && ix >= 0 && ix < IW) {
                float v = (float)in[((long)nc * IH + iy) * IW + ix];
                best = v > best ? v : best;
            }
        }
    out[idx] = (_Float16)best;
}

// ---------------------------------------------------------------- head path

// feat (15,512,8,8) f16 -> pf5 (5,512) f32 : mean over 3 S-images and 64 pixels
__global__ void k_poolfeat(const _Float16* __restrict__ feat, float* __restrict__ pf5) {
    int idx = blockIdx.x * blockDim.x + threadIdx.x;
    if (idx >= 5 * 512) return;
    int c = idx % 512, n = idx / 512;
    float s = 0.f;
    for (int si = 0; si < 3; si++) {
        const _Float16* base = feat + (((long)(n * 3 + si)) * 512 + c) * 64;
        for (int i = 0; i < 64; i++) s += (float)base[i];
    }
    pf5[idx] = s * (1.f / 192.f);
}

// yy_all (5,320) = pf5 (5,512) . [head0|head1|head2]^T (1x1 convs, no bias/BN)
__global__ void k_headgemm(const float* __restrict__ pf5,
                           const float* __restrict__ h0, const float* __restrict__ h1,
                           const float* __restrict__ h2, float* __restrict__ yy) {
    int idx = blockIdx.x * blockDim.x + threadIdx.x;
    if (idx >= 5 * 320) return;
    int col = idx % 320, n = idx / 320;
    const float* w;
    if      (col < 64)  w = h0 + (long)col * 512;
    else if (col < 192) w = h1 + (long)(col - 64) * 512;
    else                w = h2 + (long)(col - 192) * 512;
    const float* pv = pf5 + n * 512;
    float s = 0.f;
    for (int k = 0; k < 512; k++) s += pv[k] * w[k];
    yy[idx] = s;
}

// Build per-head padded ycode matrices Ych[3][16][64] f16, row r = y*oc + o
__global__ void k_ycode(const float* __restrict__ yy, _Float16* __restrict__ Ych) {
    int idx = blockIdx.x * blockDim.x + threadIdx.x;
    if (idx >= 3 * 16 * 64) return;
    int f = idx & 63, r = (idx >> 6) & 15, h = idx >> 10;
    int oc = (h == 0) ? 1 : 2;
    int cb = (h == 0) ? 0 : (h == 1 ? 64 : 192);
    float v = 0.f;
    if (r < 5 * oc) v = yy[(r / oc) * 320 + cb + (r % oc) * 64 + f];
    Ych[idx] = (_Float16)v;
}

// ---------------------------------------------------------------- einsum (WMMA + async LDS)
// per (head,x): out[x,y,o,w] = sum_f ycode[y,o,f] * xq[x,f,w]. K=64, N=16384.
// grid = (256 w-tiles of 64, 15 = head*5+x); 4 waves, each one 16-w WMMA tile.
__global__ void k_einsum(const float* __restrict__ x0, const float* __restrict__ x1,
                         const float* __restrict__ x2, const _Float16* __restrict__ Ych,
                         float* __restrict__ out) {
    const int hx = blockIdx.y, h = hx / 5, x = hx % 5;
    const float* xb = (h == 0 ? x0 : (h == 1 ? x1 : x2)) + (long)x * 64 * 16384;
    const int w0 = blockIdx.x * 64;

    __shared__ float lX[64][68];               // raw f32 tile [f][w], padded rows
    const int t = threadIdx.x;

#if defined(HAVE_ASYNC_LDS)
    // Stream 16KB tile straight into LDS: 8x global_load_async_to_lds_b128 / thread.
#pragma unroll
    for (int j = 0; j < 8; j++) {
        int idx4 = (j * 128 + t) * 4;          // float index in 64x64 tile
        int f = idx4 >> 6, wc = idx4 & 63;
        __builtin_amdgcn_global_load_async_to_lds_b128(
            (v4i*)(xb + (long)f * 16384 + w0 + wc),
            (v4i*)&lX[f][wc],
            0, 0);
    }
    wait_async0();
#else
#pragma unroll
    for (int j = 0; j < 8; j++) {
        int idx4 = (j * 128 + t) * 4;
        int f = idx4 >> 6, wc = idx4 & 63;
        *(float4*)&lX[f][wc] = *(const float4*)(xb + (long)f * 16384 + w0 + wc);
    }
#endif
    __syncthreads();

    const int wave = t >> 5, lane = t & 31;
    const int n = lane & 15, g = lane >> 4;
    const _Float16* Y = Ych + h * 16 * 64;

    v8f acc = {0.f,0.f,0.f,0.f,0.f,0.f,0.f,0.f};
#pragma unroll
    for (int c = 0; c < 2; c++) {
        int kb = c * 32;
        v16h a, b;
        const v8h* yp = (const v8h*)(Y + n * 64 + kb);
        v8h alo = yp[2 * g], ahi = yp[2 * g + 1];
#pragma unroll
        for (int e = 0; e < 8; e++) { a[e] = alo[e]; a[8 + e] = ahi[e]; }
#pragma unroll
        for (int e = 0; e < 16; e++)
            b[e] = (_Float16)lX[kb + 16 * g + e][wave * 16 + n];
        acc = __builtin_amdgcn_wmma_f32_16x16x32_f16(false, a, false, b,
                                                     (short)0, acc, false, false);
    }

    const int oc_h = (h == 0) ? 1 : 2;
    const int base = (h == 0) ? 0 : (h == 1 ? 1 : 3);
#pragma unroll
    for (int r = 0; r < 8; r++) {
        int m = r + 8 * g;
        if (m < 5 * oc_h) {
            int y = m / oc_h, o = m % oc_h;
            int w = w0 + wave * 16 + n;
            out[(((long)(x * 5 + y)) * 5 + (base + o)) * 16384 + w] = acc[r];
        }
    }
}

// ---------------------------------------------------------------- orchestration

extern "C" void kernel_launch(void* const* d_in, const int* in_sizes, int n_in,
                              void* d_out, int out_size, void* d_ws, size_t ws_size,
                              hipStream_t stream) {
    (void)in_sizes; (void)n_in; (void)out_size; (void)ws_size;
    const float* y  = (const float*)d_in[0];
    const float* x0 = (const float*)d_in[1];
    const float* x1 = (const float*)d_in[2];
    const float* x2 = (const float*)d_in[3];
    float* out = (float*)d_out;

    char* wsb = (char*)d_ws; size_t cur = 0;
    auto alloc  = [&](size_t bytes) -> char* {
        char* p = wsb + cur; cur += (bytes + 255) & ~(size_t)255; return p;
    };
    auto allocH = [&](size_t n) -> _Float16* { return (_Float16*)alloc(n * 2); };
    auto allocF = [&](size_t n) -> float*    { return (float*)alloc(n * 4); };

    // ---- activation buffers (f16)
    _Float16* inH  = allocH(15L * 3 * 256 * 256);
    _Float16* stem = allocH(15L * 64 * 127 * 127);
    _Float16* pool = allocH(15L * 64 * 64 * 64);
    _Float16* t1a  = allocH(15L * 64 * 64 * 64);
    _Float16* t1b  = allocH(15L * 64 * 64 * 64);
    _Float16* l2a  = allocH(15L * 128 * 32 * 32);
    _Float16* l2b  = allocH(15L * 128 * 32 * 32);
    _Float16* l2c  = allocH(15L * 128 * 32 * 32);
    _Float16* l3a  = allocH(15L * 256 * 16 * 16);
    _Float16* l3b  = allocH(15L * 256 * 16 * 16);
    _Float16* l3c  = allocH(15L * 256 * 16 * 16);
    _Float16* l4a  = allocH(15L * 512 * 8 * 8);
    _Float16* l4b  = allocH(15L * 512 * 8 * 8);
    _Float16* l4c  = allocH(15L * 512 * 8 * 8);
    float*    pf5  = allocF(5 * 512);
    float*    yy   = allocF(5 * 320);
    _Float16* Ych  = allocH(3 * 16 * 64);

    // ---- conv specs: {weight idx, bn-g idx, OC, IC, kernel} in d_in insertion order
    struct Spec { int wi, bni, OC, IC, K; };
    const Spec sp[20] = {
        {4,   5, 64,  3,  7},   // 0  stem (s2 p2)
        {9,  10, 64, 64,  3},   // 1  l1b0 conv1
        {14, 15, 64, 64,  3},   // 2  l1b0 conv2
        {19, 20, 64, 64,  3},   // 3  l1b1 conv1
        {24, 25, 64, 64,  3},   // 4  l1b1 conv2
        {29, 30, 128, 64, 3},   // 5  l2b0 conv1 (s2)
        {34, 35, 128,128, 3},   // 6  l2b0 conv2
        {39, 40, 128, 64, 1},   // 7  l2b0 down  (s2 p0)
        {44, 45, 128,128, 3},   // 8  l2b1 conv1
        {49, 50, 128,128, 3},   // 9  l2b1 conv2
        {54, 55, 256,128, 3},   // 10 l3b0 conv1 (s2)
        {59, 60, 256,256, 3},   // 11 l3b0 conv2
        {64, 65, 256,128, 1},   // 12 l3b0 down
        {69, 70, 256,256, 3},   // 13 l3b1 conv1
        {74, 75, 256,256, 3},   // 14 l3b1 conv2
        {79, 80, 512,256, 3},   // 15 l4b0 conv1 (s2)
        {84, 85, 512,512, 3},   // 16 l4b0 conv2
        {89, 90, 512,256, 1},   // 17 l4b0 down
        {94, 95, 512,512, 3},   // 18 l4b1 conv1
        {99,100, 512,512, 3},   // 19 l4b1 conv2
    };
    _Float16* w16[20]; float* bias[20]; int Kc[20], Kpad[20];
    for (int i = 0; i < 20; i++) {
        Kc[i]   = sp[i].IC * sp[i].K * sp[i].K;
        Kpad[i] = (Kc[i] + 31) / 32 * 32;
        w16[i]  = allocH((size_t)sp[i].OC * Kpad[i]);
        bias[i] = allocF(sp[i].OC);
    }

    // ---- weight prep
    {
        long n = 15L * 3 * 256 * 256;
        k_tohalf<<<dim3((unsigned)((n + 255) / 256)), 256, 0, stream>>>(y, inH, n);
    }
    for (int i = 0; i < 20; i++) {
        int n = sp[i].OC * Kpad[i];
        k_foldbn<<<dim3((n + 255) / 256), 256, 0, stream>>>(
            (const float*)d_in[sp[i].wi],
            (const float*)d_in[sp[i].bni],     (const float*)d_in[sp[i].bni + 1],
            (const float*)d_in[sp[i].bni + 2], (const float*)d_in[sp[i].bni + 3],
            w16[i], bias[i], sp[i].OC, Kc[i], Kpad[i]);
    }

    auto conv = [&](const _Float16* in, int s, const _Float16* res, _Float16* o,
                    int IH, int IW, int st, int pd, int relu) {
        ConvP p;
        p.in = in; p.w = w16[s]; p.bias = bias[s]; p.res = res; p.out = o;
        p.N = 15; p.IC = sp[s].IC; p.IH = IH; p.IW = IW; p.OC = sp[s].OC;
        p.KH = sp[s].K; p.KW = sp[s].K; p.st = st; p.pad = pd; p.relu = relu;
        p.OH = (IH + 2 * pd - p.KH) / st + 1;
        p.OW = (IW + 2 * pd - p.KW) / st + 1;
        p.Kc = Kc[s]; p.Kpad = Kpad[s];
        int M = p.N * p.OH * p.OW;
        dim3 grid((M + 15) / 16, p.OC / 64);
        k_conv_wmma<<<grid, 128, 0, stream>>>(p);
    };

    // ---- backbone
    conv(inH, 0, nullptr, stem, 256, 256, 2, 2, 1);               // 15x64x127x127
    {
        int total = 15 * 64 * 64 * 64;
        k_maxpool<<<dim3((total + 255) / 256), 256, 0, stream>>>(stem, pool, 15 * 64, 127, 127, 64, 64);
    }
    // layer1
    conv(pool, 1, nullptr, t1a, 64, 64, 1, 1, 1);
    conv(t1a,  2, pool,    t1b, 64, 64, 1, 1, 1);
    conv(t1b,  3, nullptr, t1a, 64, 64, 1, 1, 1);
    conv(t1a,  4, t1b,     pool,64, 64, 1, 1, 1);                  // layer1 out = pool
    // layer2
    conv(pool, 7, nullptr, l2c, 64, 64, 2, 0, 0);                  // downsample
    conv(pool, 5, nullptr, l2a, 64, 64, 2, 1, 1);
    conv(l2a,  6, l2c,     l2b, 32, 32, 1, 1, 1);
    conv(l2b,  8, nullptr, l2a, 32, 32, 1, 1, 1);
    conv(l2a,  9, l2b,     l2c, 32, 32, 1, 1, 1);                  // layer2 out = l2c
    // layer3
    conv(l2c, 12, nullptr, l3c, 32, 32, 2, 0, 0);
    conv(l2c, 10, nullptr, l3a, 32, 32, 2, 1, 1);
    conv(l3a, 11, l3c,     l3b, 16, 16, 1, 1, 1);
    conv(l3b, 13, nullptr, l3a, 16, 16, 1, 1, 1);
    conv(l3a, 14, l3b,     l3c, 16, 16, 1, 1, 1);                  // layer3 out = l3c
    // layer4
    conv(l3c, 17, nullptr, l4c, 16, 16, 2, 0, 0);
    conv(l3c, 15, nullptr, l4a, 16, 16, 2, 1, 1);
    conv(l4a, 16, l4c,     l4b, 8, 8, 1, 1, 1);
    conv(l4b, 18, nullptr, l4a, 8, 8, 1, 1, 1);
    conv(l4a, 19, l4b,     l4c, 8, 8, 1, 1, 1);                    // feat = l4c

    // ---- heads + einsum
    k_poolfeat<<<dim3(10), 256, 0, stream>>>(l4c, pf5);
    k_headgemm<<<dim3(7), 256, 0, stream>>>(pf5,
        (const float*)d_in[104], (const float*)d_in[105], (const float*)d_in[106], yy);
    k_ycode<<<dim3(12), 256, 0, stream>>>(yy, Ych);
    k_einsum<<<dim3(256, 15), 128, 0, stream>>>(x0, x1, x2, Ych, out);
}
